// MoIETransformerBlock_60696477827270
// MI455X (gfx1250) — matrix operand order repaired and checked by reading, not compiled, not run on hardware
//
#include <hip/hip_runtime.h>

// ---------------------------------------------------------------------------
// MoIE transformer block for MI455X (gfx1250), bf16 WMMA + f32 accumulate,
// TDM (tensor_load_to_lds) double-buffered staging.
// B=4, S=2048, H=768. All GEMM dims divisible by 256/128/32 -> no edge guards.
// ---------------------------------------------------------------------------

#define HD   768
#define HD3  2304                 // fused proto|gate|mu_w output width
#define SEQ  2048
#define NB   4
#define MS   (NB * SEQ)           // 8192 token rows
#define INV_SQRT_H 0.03608439182435161f  // 1/sqrt(768)

typedef __bf16 bf16;
typedef __bf16 v16bf __attribute__((ext_vector_type(16)));
typedef float  v8f   __attribute__((ext_vector_type(8)));
typedef unsigned int u32;
typedef u32 u32x4 __attribute__((ext_vector_type(4)));
typedef int i32x4 __attribute__((ext_vector_type(4)));
typedef int i32x8 __attribute__((ext_vector_type(8)));

#if __has_include(<hip/amd_detail/amd_gfx1250_TDM.h>)
#define MOIE_TDM_6ARG 1
#endif

// ------------------------------------------------------------ TDM descriptors
// D# group0: count=1, lds_addr, 57-bit global_addr, type=2 (ISA 08 §8.3)
__device__ __forceinline__ u32x4 moie_tdm_g0(unsigned long long ga, u32 lds) {
  u32x4 g0;
  g0[0] = 1u;
  g0[1] = lds;
  g0[2] = (u32)ga;
  g0[3] = (u32)((ga >> 32) & 0x01ffffffu) | (2u << 30);
  return g0;
}

// D# group1 for a [rows x 32] bf16 tile, row stride `stride` elements.
// data_size=2B; LDS pad: +4 DWORDs (16B) every 16 DWORDs (64B) -> stride-40 rows.
__device__ __forceinline__ i32x8 moie_tdm_g1(u32 rows, unsigned long long stride) {
  const u32 td0 = 32;
  i32x8 g1;
  g1[0] = (int)((1u << 16)        // data_size = 1 (2 bytes)
              | (1u << 20)        // pad_enable
              | (3u << 22)        // pad_interval code 3 -> every 16 DWORDs
              | (3u << 25));      // pad_amount  code 3 -> 4 DWORDs (16B)
  g1[1] = (int)((td0 & 0xffffu) << 16);                    // tensor_dim0.lo
  g1[2] = (int)((td0 >> 16) | ((rows & 0xffffu) << 16));   // tdim0.hi|tdim1.lo
  g1[3] = (int)((rows >> 16) | (td0 << 16));               // tdim1.hi|tile_dim0
  g1[4] = (int)(rows & 0xffffu);                           // tile_dim1, tile_dim2=0
  g1[5] = (int)(stride & 0xffffffffu);                     // dim0_stride[31:0]
  g1[6] = (int)((stride >> 32) & 0xffffu);                 // dim0_stride[47:32]
  g1[7] = 0;
  return g1;
}

__device__ __forceinline__ void moie_tdm_load(u32x4 g0, i32x8 g1) {
  i32x4 gz = {0, 0, 0, 0};
#ifdef MOIE_TDM_6ARG
  i32x8 z8 = {0, 0, 0, 0, 0, 0, 0, 0};
  __builtin_amdgcn_tensor_load_to_lds(g0, g1, gz, gz, z8, 0);
#else
  __builtin_amdgcn_tensor_load_to_lds(g0, g1, gz, gz, 0);
#endif
}

// ---------------------------------------------------------------- reductions
__device__ __forceinline__ float block_sum256(float v, float* sm) {
  int tid = threadIdx.x;
  sm[tid] = v; __syncthreads();
  #pragma unroll
  for (int s = 128; s > 0; s >>= 1) {
    if (tid < s) sm[tid] += sm[tid + s];
    __syncthreads();
  }
  float r = sm[0]; __syncthreads();
  return r;
}

__device__ __forceinline__ float block_max256(float v, float* sm) {
  int tid = threadIdx.x;
  sm[tid] = v; __syncthreads();
  #pragma unroll
  for (int s = 128; s > 0; s >>= 1) {
    if (tid < s) sm[tid] = fmaxf(sm[tid], sm[tid + s]);
    __syncthreads();
  }
  float r = sm[0]; __syncthreads();
  return r;
}

// ---------------------------------------------------------------- f32 -> bf16
__global__ void moie_cast_bf16(const float* __restrict__ in,
                               bf16* __restrict__ out, int n) {
  int i = blockIdx.x * blockDim.x + threadIdx.x;
  if (i < n) out[i] = (bf16)in[i];
}

// ---------------------------------------------------------------- WMMA GEMM
// C[m][n] = sum_k A[m][k] * Bsrc(k,n)
//   bT==0 : Bsrc is [N,K] row-major -> B(k,n) = Bsrc[n*ldb + k]   (TDM staged)
//   bT==1 : Bsrc is [K,N] row-major -> B(k,n) = Bsrc[k*ldb + n]   (manual stage)
// mode: 0 = dense; 1 = causal block-skip (QK^T); 2 = causal K-clamp (P.V)
// grid: (N/128, M/256, batches). Block = 256x128, 8 waves (4x2), wave = 64x64.
// A tile (and bT=0 B tile) stream global->LDS via TDM, double-buffered: wave 0
// issues tile i+1, all waves compute tile i, s_wait_tensorcnt + barrier.
#define BLK_M 256
#define BLK_N 128
#define BLK_K 32
#define LDSW  40   // padded LDS row stride (bf16); 80B keeps 16B alignment

__global__ __launch_bounds__(256)
void moie_gemm_bf16_wmma(const bf16* __restrict__ A, const bf16* __restrict__ B,
                         float* __restrict__ C,
                         int K, int lda, int ldb, int ldc, int bT, int mode,
                         long long sA, long long sB, long long sC) {
  __shared__ bf16 As[2][BLK_M][LDSW];
  __shared__ bf16 Ws[2][BLK_N][LDSW];   // Ws[buf][n][k]

  const int m0 = blockIdx.y * BLK_M;
  const int n0 = blockIdx.x * BLK_N;
  if (mode == 1 && n0 > m0 + BLK_M - 1) return;      // fully-masked QK block
  const int kEnd = (mode == 2 && K > m0 + BLK_M) ? (m0 + BLK_M) : K;
  const int nT = kEnd / BLK_K;

  const int z = blockIdx.z;
  A += (long long)z * sA;
  B += (long long)z * sB;
  C += (long long)z * sC;

  const int tid  = threadIdx.x;
  const int lane = tid & 31;
  const int wid  = tid >> 5;
  const int wm   = wid >> 1;      // 0..3  -> 64-row slab
  const int wn   = wid & 1;       // 0..1  -> 64-col slab
  const int lh   = lane & 15;
  const int hi   = (lane >> 4) & 1;
  const bool issuer = (wid == 0);

  const unsigned long long gaA =
      (unsigned long long)(uintptr_t)(A + (long long)m0 * lda);
  const unsigned long long gaB =
      (unsigned long long)(uintptr_t)(B + (long long)n0 * ldb);   // bT=0 only
  const u32 ldsA[2] = { (u32)(uintptr_t)&As[0][0][0], (u32)(uintptr_t)&As[1][0][0] };
  const u32 ldsB[2] = { (u32)(uintptr_t)&Ws[0][0][0], (u32)(uintptr_t)&Ws[1][0][0] };
  const i32x8 g1A = moie_tdm_g1(BLK_M, (unsigned long long)(u32)lda);
  const i32x8 g1B = moie_tdm_g1(BLK_N, (unsigned long long)(u32)ldb);

  // manual transpose-staging of B for bT=1: Ws[buf][n][k] = B[(k0+k)*ldb + n0+n]
  auto stageBT = [&](int k0, int buf) {
    #pragma unroll
    for (int it = 0; it < 2; ++it) {
      int idx = tid + it * 256;          // (k 0..31) x (n-chunk 0..15)
      int kk  = idx >> 4;
      int c8  = (idx & 15) * 8;
      const bf16* src = B + (long long)(k0 + kk) * ldb + n0 + c8;
      bf16 tmp[8];
      *(uint4*)tmp = *(const uint4*)src;
      #pragma unroll
      for (int t = 0; t < 8; ++t) Ws[buf][c8 + t][kk] = tmp[t];
    }
  };

  // ---------------- prologue: tile 0 into buffer 0 ----------------
  if (issuer) {
    moie_tdm_load(moie_tdm_g0(gaA, ldsA[0]), g1A);
    if (!bT) moie_tdm_load(moie_tdm_g0(gaB, ldsB[0]), g1B);
  }
  if (bT) stageBT(0, 0);
  if (issuer) __builtin_amdgcn_s_wait_tensorcnt(0);
  __syncthreads();

  const v8f vzero = {0.f, 0.f, 0.f, 0.f, 0.f, 0.f, 0.f, 0.f};
  v8f acc[4][4];
  #pragma unroll
  for (int i = 0; i < 4; ++i)
    #pragma unroll
    for (int j = 0; j < 4; ++j) acc[i][j] = vzero;

  union FragU { v16bf v; uint4 u[2]; };

  for (int i = 0; i < nT; ++i) {
    const int cur = i & 1;
    const int nxt = cur ^ 1;

    // prefetch tile i+1 while computing tile i
    if (i + 1 < nT) {
      if (issuer) {
        unsigned long long adv = (unsigned long long)(i + 1) * (BLK_K * 2);
        moie_tdm_load(moie_tdm_g0(gaA + adv, ldsA[nxt]), g1A);
        if (!bT) moie_tdm_load(moie_tdm_g0(gaB + adv, ldsB[nxt]), g1B);
      }
      if (bT) stageBT((i + 1) * BLK_K, nxt);
    }

    // --- per-lane fragments (ISA 7.12.2 layouts) ---------------------------
    v16bf af[4], bfv[4];
    #pragma unroll
    for (int mf = 0; mf < 4; ++mf) {
      int row = wm * 64 + mf * 16 + lh;
      FragU fu;
      fu.u[0] = *(const uint4*)&As[cur][row][hi * 8];        // K 0-7   / 8-15
      fu.u[1] = *(const uint4*)&As[cur][row][16 + hi * 8];   // K 16-23 / 24-31
      af[mf] = fu.v;
    }
    #pragma unroll
    for (int nf = 0; nf < 4; ++nf) {
      int col = wn * 64 + nf * 16 + lh;
      FragU fu;
      fu.u[0] = *(const uint4*)&Ws[cur][col][hi * 8];        // K 0-7   / 8-15
      fu.u[1] = *(const uint4*)&Ws[cur][col][16 + hi * 8];   // K 16-23 / 24-31
      bfv[nf] = fu.v;
    }

    #pragma unroll
    for (int mf = 0; mf < 4; ++mf)
      #pragma unroll
      for (int nf = 0; nf < 4; ++nf)
        acc[mf][nf] = __builtin_amdgcn_wmma_f32_16x16x32_bf16(
            false, af[mf], false, bfv[nf], (short)0, acc[mf][nf], false, false);

    if (issuer && i + 1 < nT) __builtin_amdgcn_s_wait_tensorcnt(0);
    __syncthreads();
  }

  // --- store C (lane<16 -> M=r, lane>=16 -> M=r+8) --------------------------
  #pragma unroll
  for (int mf = 0; mf < 4; ++mf) {
    #pragma unroll
    for (int nf = 0; nf < 4; ++nf) {
      int row0 = m0 + wm * 64 + mf * 16 + hi * 8;
      int col  = n0 + wn * 64 + nf * 16 + lh;
      #pragma unroll
      for (int r = 0; r < 8; ++r)
        C[(long long)(row0 + r) * ldc + col] = acc[mf][nf][r];
    }
  }
}

// ---------------------------------------------------------------- LayerNorm
__global__ __launch_bounds__(256)
void moie_layernorm(const float* __restrict__ x, const float* __restrict__ w,
                    const float* __restrict__ b, float* __restrict__ of,
                    bf16* __restrict__ ob) {
  __shared__ float sm[256];
  long long row = blockIdx.x;
  int tid = threadIdx.x;
  const float* xr = x + row * HD;
  float v[3], s = 0.f;
  #pragma unroll
  for (int t = 0; t < 3; ++t) { v[t] = xr[tid + t * 256]; s += v[t]; }
  float mu = block_sum256(s, sm) * (1.0f / HD);
  float ss = 0.f;
  #pragma unroll
  for (int t = 0; t < 3; ++t) { float d = v[t] - mu; ss += d * d; }
  float inv = rsqrtf(block_sum256(ss, sm) * (1.0f / HD) + 1e-5f);
  #pragma unroll
  for (int t = 0; t < 3; ++t) {
    int c = tid + t * 256;
    float y = (v[t] - mu) * inv * w[c] + b[c];
    of[row * HD + c] = y;
    ob[row * HD + c] = (bf16)y;
  }
}

// -------------------------------------------------- SPL gate-branch epilogue
// cmg3 is the fused [MS, 2304] GEMM output: cols 0-767 match(proto),
// 768-1535 gate_logit, 1536-2303 comp(mu_w).
// out = [res +] maybe_relu( (inp + comp + mu_b) * mas(match*scale - mas(gate)) )
__global__ __launch_bounds__(256)
void moie_spl_epilogue(const float* __restrict__ inp, const float* __restrict__ cmg3,
                       const float* __restrict__ mu_b, const float* __restrict__ res,
                       float* __restrict__ of, bf16* __restrict__ ob, int do_relu) {
  __shared__ float sm[256];
  long long base  = (long long)blockIdx.x * HD;
  long long base3 = (long long)blockIdx.x * HD3;
  int tid = threadIdx.x;
  float gl[3], mt[3], rt[3];
  float lm = 0.f;
  #pragma unroll
  for (int t = 0; t < 3; ++t) {
    int c = tid + t * 256;
    mt[t] = cmg3[base3 + c];
    gl[t] = cmg3[base3 + HD + c];
    lm = fmaxf(lm, fabsf(gl[t]));
  }
  float m1 = block_max256(lm, sm);
  lm = 0.f;
  #pragma unroll
  for (int t = 0; t < 3; ++t) {
    float g = fmaxf(gl[t] / (m1 + 1e-9f), 0.f);   // mas(gate_logit)
    rt[t] = mt[t] * INV_SQRT_H - g;               // routing (GAIN=1)
    lm = fmaxf(lm, fabsf(rt[t]));
  }
  float m2 = block_max256(lm, sm);
  #pragma unroll
  for (int t = 0; t < 3; ++t) {
    int c = tid + t * 256;
    float gate = fmaxf(rt[t] / (m2 + 1e-9f), 0.f);
    float val = (inp[base + c] + cmg3[base3 + 2 * HD + c] + mu_b[c]) * gate;
    if (do_relu) val = fmaxf(val, 0.f);
    if (res) val += res[base + c];
    if (of) of[base + c] = val;
    if (ob) ob[base + c] = (bf16)val;
  }
}

// ---------------------------------------------------------------- RoPE
__global__ __launch_bounds__(256)
void moie_rope(const float* __restrict__ q, const float* __restrict__ k,
               const float* __restrict__ cosT, const float* __restrict__ sinT,
               bf16* __restrict__ qo, bf16* __restrict__ ko) {
  long long row = blockIdx.x;
  int tid = threadIdx.x;
  int s = (int)(row % SEQ);
  long long base = row * HD;
  const int half = HD / 2;
  #pragma unroll
  for (int t = 0; t < 3; ++t) {
    int c = tid + t * 256;
    float cs = cosT[(long long)s * HD + c];
    float sn = sinT[(long long)s * HD + c];
    float qr = (c < half) ? -q[base + c + half] : q[base + c - half];
    float kr = (c < half) ? -k[base + c + half] : k[base + c - half];
    qo[base + c] = (bf16)(q[base + c] * cs + qr * sn);
    ko[base + c] = (bf16)(k[base + c] * cs + kr * sn);
  }
}

// ------------------------------------------------------- masked softmax rows
__global__ __launch_bounds__(256)
void moie_softmax(const float* __restrict__ scores, bf16* __restrict__ probs) {
  __shared__ float sm[256];
  long long row = blockIdx.x;                 // b*SEQ + q
  int tid = threadIdx.x;
  int qi = (int)(row % SEQ);
  const float* sr = scores + row * SEQ;
  bf16* pr = probs + row * SEQ;
  float v[8], lm = -3.402823e38f;
  #pragma unroll
  for (int t = 0; t < 8; ++t) {
    int c = tid + t * 256;
    v[t] = (c <= qi) ? sr[c] * INV_SQRT_H : -3.402823e38f;
    lm = fmaxf(lm, v[t]);
  }
  float m = block_max256(lm, sm);
  float ls = 0.f;
  #pragma unroll
  for (int t = 0; t < 8; ++t) {
    int c = tid + t * 256;
    float e = (c <= qi) ? __expf(v[t] - m) : 0.f;
    v[t] = e; ls += e;
  }
  float idn = 1.f / block_sum256(ls, sm);
  #pragma unroll
  for (int t = 0; t < 8; ++t) pr[tid + t * 256] = (bf16)(v[t] * idn);
}

// ---------------------------------------------------------------------------
extern "C" void kernel_launch(void* const* d_in, const int* in_sizes, int n_in,
                              void* d_out, int out_size, void* d_ws, size_t ws_size,
                              hipStream_t stream) {
  (void)in_sizes; (void)n_in; (void)out_size; (void)ws_size;

  const float* x     = (const float*)d_in[0];
  const float* cosT  = (const float*)d_in[1];
  const float* sinT  = (const float*)d_in[2];
  const float* ln1_w = (const float*)d_in[3];
  const float* ln1_b = (const float*)d_in[4];
  const float* ln2_w = (const float*)d_in[5];
  const float* ln2_b = (const float*)d_in[6];
  // layers l=0..5 (q,k,v,o,f1,f2): d_in[7+4l] = proto, +1 gate, +2 mu_w, +3 mu_b

  // ---------------- workspace layout ----------------
  char* base = (char*)d_ws;
  size_t off = 0;
  auto take = [&](size_t bytes) -> char* {
    char* p = base + off;
    off += (bytes + 255) & ~(size_t)255;
    return p;
  };
  const size_t WN   = (size_t)HD * HD;            // elems per weight matrix
  const size_t MSH2 = (size_t)MS * HD * 2;        // bf16 activation bytes
  const size_t MSH4 = (size_t)MS * HD * 4;        // f32 activation bytes
  const size_t SC4  = (size_t)NB * SEQ * SEQ * 4; // f32 scores bytes
  const size_t CMG  = (size_t)MS * HD3 * 4;       // fused match|gate|comp

  bf16* Wbf = (bf16*)take(WN * 2 * 18);   // 18 bf16 weight matrices
  bf16* Abf = (bf16*)take(MSH2);          // staged GEMM input (reused)
  bf16* Qbf = (bf16*)take(MSH2);
  bf16* Kbf = (bf16*)take(MSH2);
  bf16* Vbf = (bf16*)take(MSH2);
  char* cmg = take(CMG > SC4 ? CMG : SC4);        // scores overlay cmg
  float* cmg3   = (float*)cmg;                    // [MS, 2304]
  float* scores = (float*)cmg;                    // disjoint lifetime
  bf16*  probs  = (bf16*)take((size_t)NB * SEQ * SEQ * 2);
  float* F0 = (float*)take(MSH4);   // ln1_out   -> later h_act
  float* F1 = (float*)take(MSH4);   // q_f32     -> later attn_out
  float* F2 = (float*)take(MSH4);   // k_f32     -> later ln2_out
  float* F3 = (float*)take(MSH4);   // x1 = x + m_o

  // ---------------- cast all 18 weight matrices to bf16 ----------------
  {
    int n = (int)WN;
    int g = (n + 255) / 256;
    for (int l = 0; l < 6; ++l)
      for (int s = 0; s < 3; ++s)  // 0=proto,1=gate,2=mu_w (contiguous per layer)
        moie_cast_bf16<<<g, 256, 0, stream>>>(
            (const float*)d_in[7 + l * 4 + s], Wbf + (size_t)(l * 3 + s) * WN, n);
  }

  auto gemm = [&](const bf16* A, const bf16* B, float* C, int M, int N, int K,
                  int lda, int ldb, int ldc, int bT, int mode, int zb,
                  long long sA, long long sB, long long sC) {
    dim3 g(N / BLK_N, M / BLK_M, zb);
    moie_gemm_bf16_wmma<<<g, 256, 0, stream>>>(A, B, C, K, lda, ldb, ldc,
                                               bT, mode, sA, sB, sC);
  };

  // Fused SPL: one GEMM over [proto|gate|mu_w] (contiguous, ldb=HD) -> cmg3
  auto spl = [&](int l, const float* inp_f32, const bf16* inp_bf,
                 const float* res, float* of, bf16* ob, int relu) {
    const float* mub = (const float*)d_in[7 + l * 4 + 3];
    const bf16* Wp = Wbf + (size_t)(l * 3) * WN;
    gemm(inp_bf, Wp, cmg3, MS, HD3, HD, HD, HD, HD3, 0, 0, 1, 0, 0, 0);
    moie_spl_epilogue<<<MS, 256, 0, stream>>>(inp_f32, cmg3, mub, res, of, ob, relu);
  };

  // 1) LN1: x -> F0 (f32) + Abf (bf16)
  moie_layernorm<<<MS, 256, 0, stream>>>(x, ln1_w, ln1_b, F0, Abf);

  // 2) q, k, v SPL branches
  spl(0, F0, Abf, nullptr, F1, nullptr, 0);   // q -> F1
  spl(1, F0, Abf, nullptr, F2, nullptr, 0);   // k -> F2
  spl(2, F0, Abf, nullptr, nullptr, Vbf, 0);  // v -> bf16 directly

  // 3) RoPE -> bf16 Q/K
  moie_rope<<<MS, 256, 0, stream>>>(F1, F2, cosT, sinT, Qbf, Kbf);

  // 4) scores = Q K^T (masked blocks skipped), softmax -> bf16 probs
  gemm(Qbf, Kbf, scores, SEQ, SEQ, HD, HD, HD, SEQ, 0, 1, NB,
       (long long)SEQ * HD, (long long)SEQ * HD, (long long)SEQ * SEQ);
  moie_softmax<<<MS, 256, 0, stream>>>(scores, probs);

  // 5) attn_out = P V (bT=1, causal K-clamp) -> F1
  gemm(probs, Vbf, F1, SEQ, HD, SEQ, SEQ, HD, HD, 1, 2, NB,
       (long long)SEQ * SEQ, (long long)SEQ * HD, (long long)SEQ * HD);
  moie_cast_bf16<<<(MS * HD + 255) / 256, 256, 0, stream>>>(F1, Abf, MS * HD);

  // 6) o-layer SPL + residual: x1 = x + m_o -> F3
  spl(3, F1, Abf, x, F3, nullptr, 0);

  // 7) LN2: F3 -> F2 (f32) + Abf (bf16)
  moie_layernorm<<<MS, 256, 0, stream>>>(F3, ln2_w, ln2_b, F2, Abf);

  // 8) f1 SPL + relu: -> F0 (f32) + Abf (bf16)   (Abf rewritten post-GEMM)
  spl(4, F2, Abf, nullptr, F0, Abf, 1);

  // 9) f2 SPL + residual: out = F3 + m2 -> d_out
  spl(5, F0, Abf, F3, (float*)d_out, nullptr, 0);
}
